// FullAttention_52372831208134
// MI455X (gfx1250) — matrix operand order repaired and checked by reading, not compiled, not run on hardware
//
#include <hip/hip_runtime.h>
#include <stdint.h>

// Problem constants (from reference)
#define N_B 4
#define L_Q 2048
#define S_K 2048
#define H_N 8
#define D_H 64

#define BM 64          // query rows per workgroup (4 waves x 16)
#define BN 64          // keys per S-iteration
#define RSTRIDE 68     // f32 row stride of TDM-landed raw tiles (64 + 4 pad DW)
#define KSTRIDE 68     // f16 row stride for lds_k  (64 + 4 pad)
#define VSTRIDE 68     // f16 row stride for lds_vt (64 + 4 pad)
#define PSTRIDE 66     // f16 row stride for lds_p  (64 + 2 pad)

typedef _Float16 v16h __attribute__((ext_vector_type(16)));
typedef _Float16 v2h  __attribute__((ext_vector_type(2)));
typedef float    v8f  __attribute__((ext_vector_type(8)));
typedef unsigned int v4u __attribute__((ext_vector_type(4)));
typedef int      v8i  __attribute__((ext_vector_type(8)));
typedef int      v4i  __attribute__((ext_vector_type(4)));

union U16H { v16h v; _Float16 h[16]; };
union U8F  { v8f  v; float    f[8];  };

// A-matrix (16x32 f16) K-offset for VGPR v on lane-half 'hl' (ISA 7.12.2)
__device__ __forceinline__ int koffA(int v, int hl) {
  return ((v < 4) ? 0 : 16) + (v & 3) * 2 + hl * 8;
}
// B-matrix (32x16 f16) K-offset: same K order as the A-matrix
__device__ __forceinline__ int koffB(int v, int hl) {
  return ((v < 4) ? 0 : 16) + (v & 3) * 2 + hl * 8;
}

// Issue one TDM 2D tile load: 64 rows x 64 f32, global row stride H_N*D_H,
// landing in LDS with RSTRIDE-dword rows (pad_interval=64DW, pad_amount=4DW).
__device__ __forceinline__ void tdm_load_tile(uint32_t lds_byte_off,
                                              const float* gsrc) {
  const uint64_t ga = (uint64_t)(uintptr_t)gsrc;
  v4u g0;
  g0[0] = 1u;                                   // count=1, user descriptor
  g0[1] = lds_byte_off;                         // lds_addr
  g0[2] = (uint32_t)ga;                         // global_addr[31:0]
  g0[3] = (uint32_t)((ga >> 32) & 0x1FFFFFFu)   // global_addr[56:32]
          | (2u << 30);                         // type=2 ("image")
  v8i g1;
  g1[0] = (int)((2u << 16)    // data_size = 4 bytes
              | (1u << 20)    // pad_enable
              | (5u << 22)    // pad_interval: 64 DWORDs
              | (3u << 25));  // pad_amount: 4 DWORDs
  g1[1] = (int)(64u << 16);   // tensor_dim0[15:0] = 64  (bits 63:48)
  g1[2] = (int)(64u << 16);   // tensor_dim1[15:0] = 64  (bits 95:80)
  g1[3] = (int)(64u << 16);   // tile_dim0 = 64          (bits 127:112)
  g1[4] = 64;                 // tile_dim1 = 64          (bits 143:128)
  g1[5] = (int)(H_N * D_H);   // tensor_dim0_stride = 512 elements
  g1[6] = 0;
  g1[7] = 0;
  v4i z4; z4[0] = 0; z4[1] = 0; z4[2] = 0; z4[3] = 0;   // 2D: groups 2/3 unused
  v8i z8;
  #pragma unroll
  for (int i = 0; i < 8; ++i) z8[i] = 0;
  __builtin_amdgcn_tensor_load_to_lds(g0, g1, z4, z4, z8, 0);
}

__global__ __launch_bounds__(128, 1) void fullattn_fwd(
    const float* __restrict__ Q, const float* __restrict__ K,
    const float* __restrict__ V, const uint8_t* __restrict__ QM,
    const uint8_t* __restrict__ KM, float* __restrict__ O)
{
  __shared__ __align__(16) float lds_kraw[2][BN * RSTRIDE];  // TDM K landing
  __shared__ __align__(16) float lds_vraw[2][BN * RSTRIDE];  // TDM V landing
  __shared__ _Float16 lds_k [BN  * KSTRIDE];    // K block, [s][d], f16
  __shared__ _Float16 lds_vt[D_H * VSTRIDE];    // V block transposed, [d][s]
  __shared__ _Float16 lds_p [4 * 16 * PSTRIDE]; // per-wave P scratch, [m][key]

  const int tid  = threadIdx.x;
  const int wave = tid >> 5;
  const int lane = tid & 31;
  const int hl   = lane >> 4;   // lane half (0: lanes 0-15, 1: lanes 16-31)
  const int nn   = lane & 15;

  const int lblocks = L_Q / BM;
  int bid = blockIdx.x;
  const int lb = bid % lblocks; bid /= lblocks;
  const int h  = bid % H_N;     bid /= H_N;
  const int n  = bid;

  const int l0 = lb * BM + wave * 16;   // first query row of this wave
  const size_t kvhead = ((size_t)n * S_K * H_N + h) * D_H;  // + s*H_N*D_H

  // ---- prefetch S-block 0 via TDM (wave 0 only; EXEC is ignored by TDM) ----
  if (wave == 0) {
    tdm_load_tile((uint32_t)(uintptr_t)&lds_kraw[0][0], K + kvhead);
    tdm_load_tile((uint32_t)(uintptr_t)&lds_vraw[0][0], V + kvhead);
  }

  // ---- Q tile -> A-layout f16 regs, scaled by 1/sqrt(D) ----
  U16H aq0, aq1;   // K-chunks d=0..31 and d=32..63
  {
    const float sc = 0.125f;
    const size_t qb = ((((size_t)n * L_Q + (l0 + nn)) * H_N) + h) * D_H;
    #pragma unroll
    for (int v = 0; v < 8; ++v) {
      const int k0 = koffA(v, hl);
      float2 t0 = *(const float2*)(Q + qb + k0);
      float2 t1 = *(const float2*)(Q + qb + 32 + k0);
      aq0.h[2*v]   = (_Float16)(t0.x * sc);
      aq0.h[2*v+1] = (_Float16)(t0.y * sc);
      aq1.h[2*v]   = (_Float16)(t1.x * sc);
      aq1.h[2*v+1] = (_Float16)(t1.y * sc);
    }
  }

  // q-mask bit per C/D-layout row (row m = v + 8*hl)
  unsigned qmbits = 0;
  #pragma unroll
  for (int v = 0; v < 8; ++v)
    qmbits |= (QM[(size_t)n * L_Q + l0 + v + 8 * hl] ? 1u : 0u) << v;

  U8F o[4];                       // O accumulator: 4 d-tiles of 16x16 f32
  #pragma unroll
  for (int j = 0; j < 4; ++j)
    #pragma unroll
    for (int v = 0; v < 8; ++v) o[j].f[v] = 0.0f;

  float mrow[8], lrow[8];
  #pragma unroll
  for (int v = 0; v < 8; ++v) { mrow[v] = -__builtin_inff(); lrow[v] = 0.0f; }

  _Float16* pw = &lds_p[wave * 16 * PSTRIDE];

  const int NBLK = S_K / BN;
  for (int i = 0; i < NBLK; ++i) {
    const int s0 = i * BN;
    const int buf = i & 1;

    // ---- TDM completion for block i, then publish to the workgroup ----
    if (wave == 0) __builtin_amdgcn_s_wait_tensorcnt(0);
    __syncthreads();   // raw[buf] valid for all; raw[buf^1] free to overwrite

    // ---- prefetch block i+1 into the other raw buffer ----
    if (wave == 0 && i + 1 < NBLK) {
      const size_t gb = kvhead + (size_t)(s0 + BN) * H_N * D_H;
      tdm_load_tile((uint32_t)(uintptr_t)&lds_kraw[buf ^ 1][0], K + gb);
      tdm_load_tile((uint32_t)(uintptr_t)&lds_vraw[buf ^ 1][0], V + gb);
    }

    // ---- convert raw f32 -> compact f16 (K natural, V transposed) ----
    {
      const float* kr = &lds_kraw[buf][0];
      const float* vr = &lds_vraw[buf][0];
      #pragma unroll
      for (int it = 0; it < 8; ++it) {
        const int task = it * 128 + tid;
        const int sl = task >> 4;          // key row within block
        const int dg = (task & 15) * 4;    // d group of 4
        float4 kk = *(const float4*)(kr + sl * RSTRIDE + dg);
        float4 vv = *(const float4*)(vr + sl * RSTRIDE + dg);
        v2h k01 = { (_Float16)kk.x, (_Float16)kk.y };
        v2h k23 = { (_Float16)kk.z, (_Float16)kk.w };
        *(v2h*)&lds_k[sl * KSTRIDE + dg]     = k01;
        *(v2h*)&lds_k[sl * KSTRIDE + dg + 2] = k23;
        lds_vt[(dg + 0) * VSTRIDE + sl] = (_Float16)vv.x;
        lds_vt[(dg + 1) * VSTRIDE + sl] = (_Float16)vv.y;
        lds_vt[(dg + 2) * VSTRIDE + sl] = (_Float16)vv.z;
        lds_vt[(dg + 3) * VSTRIDE + sl] = (_Float16)vv.w;
      }
    }
    __syncthreads();

    // ---- scores: S = (Q/sqrt(D)) . K^T : 4 tiles 16x16, 2 K-chunks each ----
    U8F st[4];
    #pragma unroll
    for (int g = 0; g < 4; ++g) {
      U16H b0, b1;
      const _Float16* krow = &lds_k[(g * 16 + nn) * KSTRIDE];
      #pragma unroll
      for (int v = 0; v < 8; ++v) {
        const int k0 = koffB(v, hl);
        v2h p0 = *(const v2h*)(krow + k0);
        v2h p1 = *(const v2h*)(krow + 32 + k0);
        b0.h[2*v] = p0.x; b0.h[2*v+1] = p0.y;
        b1.h[2*v] = p1.x; b1.h[2*v+1] = p1.y;
      }
      v8f c = {};
      c = __builtin_amdgcn_wmma_f32_16x16x32_f16(false, aq0.v, false, b0.v,
                                                 (short)0, c, false, false);
      c = __builtin_amdgcn_wmma_f32_16x16x32_f16(false, aq1.v, false, b1.v,
                                                 (short)0, c, false, false);
      st[g].v = c;
    }

    // ---- mask (matches reference: -inf where !(q_mask & kv_mask)) ----
    #pragma unroll
    for (int g = 0; g < 4; ++g) {
      const bool km = KM[(size_t)n * S_K + s0 + g * 16 + nn] != 0;
      #pragma unroll
      for (int v = 0; v < 8; ++v) {
        const bool valid = km && ((qmbits >> v) & 1u);
        st[g].f[v] = valid ? st[g].f[v] : -__builtin_inff();
      }
    }

    // ---- online softmax (rows live in 16-lane halves -> width-16 shfl) ----
    float alpha[8];
    #pragma unroll
    for (int v = 0; v < 8; ++v) {
      float mx = fmaxf(fmaxf(st[0].f[v], st[1].f[v]),
                       fmaxf(st[2].f[v], st[3].f[v]));
      #pragma unroll
      for (int off = 1; off < 16; off <<= 1)
        mx = fmaxf(mx, __shfl_xor(mx, off, 16));
      const float mnew  = fmaxf(mrow[v], mx);
      const float a     = (mrow[v] == -__builtin_inff())
                            ? 0.0f : __expf(mrow[v] - mnew);
      const float msafe = (mnew == -__builtin_inff()) ? 0.0f : mnew;
      float rs = 0.0f;
      #pragma unroll
      for (int g = 0; g < 4; ++g) {
        const float p = __expf(st[g].f[v] - msafe);
        st[g].f[v] = p;
        rs += p;
      }
      #pragma unroll
      for (int off = 1; off < 16; off <<= 1)
        rs += __shfl_xor(rs, off, 16);
      lrow[v]  = lrow[v] * a + rs;
      mrow[v]  = mnew;
      alpha[v] = a;
    }
    #pragma unroll
    for (int j = 0; j < 4; ++j)
      #pragma unroll
      for (int v = 0; v < 8; ++v)
        o[j].f[v] *= alpha[v];

    // ---- C/D-layout P -> f16 -> per-wave LDS -> reload in A layout ----
    #pragma unroll
    for (int g = 0; g < 4; ++g)
      #pragma unroll
      for (int v = 0; v < 8; ++v)
        pw[(v + 8 * hl) * PSTRIDE + g * 16 + nn] = (_Float16)st[g].f[v];

    U16H pa0, pa1;
    {
      const _Float16* pr = pw + nn * PSTRIDE;
      #pragma unroll
      for (int v = 0; v < 8; ++v) {
        const int k0 = koffA(v, hl);
        v2h p0 = *(const v2h*)(pr + k0);
        v2h p1 = *(const v2h*)(pr + 32 + k0);
        pa0.h[2*v] = p0.x; pa0.h[2*v+1] = p0.y;
        pa1.h[2*v] = p1.x; pa1.h[2*v+1] = p1.y;
      }
    }

    // ---- O += P x V ----
    #pragma unroll
    for (int j = 0; j < 4; ++j) {
      U16H b0, b1;
      const _Float16* vrow = &lds_vt[(j * 16 + nn) * VSTRIDE];
      #pragma unroll
      for (int v = 0; v < 8; ++v) {
        const int k0 = koffB(v, hl);
        v2h p0 = *(const v2h*)(vrow + k0);
        v2h p1 = *(const v2h*)(vrow + 32 + k0);
        b0.h[2*v] = p0.x; b0.h[2*v+1] = p0.y;
        b1.h[2*v] = p1.x; b1.h[2*v+1] = p1.y;
      }
      o[j].v = __builtin_amdgcn_wmma_f32_16x16x32_f16(false, pa0.v, false, b0.v,
                                                      (short)0, o[j].v, false, false);
      o[j].v = __builtin_amdgcn_wmma_f32_16x16x32_f16(false, pa1.v, false, b1.v,
                                                      (short)0, o[j].v, false, false);
    }
    __syncthreads();
  }

  // ---- normalize by row sum and store (coalesced 16-lane runs of d) ----
  #pragma unroll
  for (int v = 0; v < 8; ++v) {
    const float inv = 1.0f / lrow[v];
    const size_t ob =
        ((((size_t)n * L_Q + (l0 + v + 8 * hl)) * H_N) + h) * D_H + nn;
    #pragma unroll
    for (int j = 0; j < 4; ++j)
      O[ob + j * 16] = o[j].f[v] * inv;
  }
}

extern "C" void kernel_launch(void* const* d_in, const int* in_sizes, int n_in,
                              void* d_out, int out_size, void* d_ws, size_t ws_size,
                              hipStream_t stream) {
  const float*   q  = (const float*)d_in[0];
  const float*   k  = (const float*)d_in[1];
  const float*   v  = (const float*)d_in[2];
  const uint8_t* qm = (const uint8_t*)d_in[3];
  const uint8_t* km = (const uint8_t*)d_in[4];
  float* out = (float*)d_out;

  dim3 grid(N_B * H_N * (L_Q / BM));   // 1024 workgroups
  dim3 block(128);                     // 4 wave32
  hipLaunchKernelGGL(fullattn_fwd, grid, block, 0, stream, q, k, v, qm, km, out);
}
